// MPNEncoder_83202106458599
// MI455X (gfx1250) — matrix operand order repaired and checked by reading, not compiled, not run on hardware
//
#include <hip/hip_runtime.h>
#include <hip/hip_bf16.h>

#define HID   256
#define NATOMS 100000
#define NBONDS 250000
#define NMOLS  2000
#define MAXNB  6
#define AFD    133
#define BFD    147
#define OFD    389   // AFD + HID

typedef __attribute__((ext_vector_type(16))) __bf16 v16bf;
typedef __attribute__((ext_vector_type(8)))  float  v8f;

union Frag { unsigned int u[8]; uint4 q[2]; v16bf v; };

__device__ __forceinline__ unsigned short f2bf(float x) {
  unsigned int u = __float_as_uint(x);
  u += 0x7FFFu + ((u >> 16) & 1u);           // round-to-nearest-even
  return (unsigned short)(u >> 16);
}
__device__ __forceinline__ unsigned int pack2bf(float x0, float x1) {
  return (unsigned int)f2bf(x0) | ((unsigned int)f2bf(x1) << 16);
}

// A fragment (16x32 bf16, ISA 7.12.2): lane m=l%16, half=l>>4.
// dwords {base..base+3} (K=8h..8h+7) and {base+8..base+11} (K=16+8h..16+8h+7)
// -> two ds_load_b128; pitch P with P%4==0 and (P/4) odd => conflict-free.
__device__ __forceinline__ void load_a_frag(const unsigned int* Alds, int P, int lane,
                                            int kstep, Frag& a) {
  const unsigned int* base = Alds + (lane & 15) * P + kstep * 16 + 4 * (lane >> 4);
  a.q[0] = *(const uint4*)(base);
  a.q[1] = *(const uint4*)(base + 8);
}

// B pre-packed as [kstep][ntile][grp][lane][4dw] -> two fully-coalesced global_load_b128
__device__ __forceinline__ void load_b_frag(const unsigned int* __restrict__ Bp, int nt,
                                            int kstep, int lane, Frag& b) {
  const uint4* p = (const uint4*)Bp + ((kstep * 16 + nt) * 2) * 32 + lane;
  b.q[0] = p[0];
  b.q[1] = p[32];
}

__device__ __forceinline__ v8f wmma_bf16(const Frag& a, const Frag& b, v8f c) {
  return __builtin_amdgcn_wmma_f32_16x16x32_bf16(false, a.v, false, b.v,
                                                 (short)0, c, false, false);
}

// ---------------- weight packing: W[K][256] f32 -> WMMA B-fragment bf16 layout ------------
__global__ __launch_bounds__(256) void k_pack(const float* __restrict__ W, int K, int total,
                                              unsigned int* __restrict__ out) {
  int idx = blockIdx.x * 256 + threadIdx.x;
  if (idx >= total) return;
  int j = idx & 3, lane = (idx >> 2) & 31, g = (idx >> 7) & 1;
  int nt = (idx >> 8) & 15, t = idx >> 12;
  int v = g * 4 + j;
  int n = nt * 16 + (lane & 15);
  int k = t * 32 + 16 * g + 8 * (lane >> 4) + 2 * j;  // B: half h, g=0 K=8h..8h+7, g=1 K=16+8h..16+8h+7
  float x0 = (k     < K) ? W[(long)k       * HID + n] : 0.f;
  float x1 = (k + 1 < K) ? W[(long)(k + 1) * HID + n] : 0.f;
  out[idx] = pack2bf(x0, x1);
}

// ---------------- GEMM 1: inp = f_bonds @ W_i ; msg = relu(inp)  (K 147 -> 160) ----------
__global__ __launch_bounds__(256) void k_gemm_inp(const float* __restrict__ f_bonds,
    const unsigned int* __restrict__ Wp, float* __restrict__ inp, float* __restrict__ msg) {
  constexpr int KP = 160, P = KP / 2 + 4;       // 84: P%4==0, P/4 odd
  __shared__ unsigned int Alds[16 * P];
  const int t = threadIdx.x, row = t >> 4, kb = t & 15;
  const long bond0 = (long)blockIdx.x * 16;
  const float* src = f_bonds + (bond0 + row) * BFD;
#pragma unroll
  for (int i = 0; i < KP / 32; ++i) {           // 5 dword-pairs per thread
    int d = kb + i * 16, k = 2 * d;
    float x0 = (k     < BFD) ? src[k]     : 0.f;
    float x1 = (k + 1 < BFD) ? src[k + 1] : 0.f;
    Alds[row * P + d] = pack2bf(x0, x1);
  }
  __syncthreads();
  const int wave = t >> 5, lane = t & 31, half = lane >> 4;
  v8f acc0 = {0.f,0.f,0.f,0.f,0.f,0.f,0.f,0.f};
  v8f acc1 = acc0;
#pragma unroll
  for (int s = 0; s < KP / 32; ++s) {
    Frag a;  load_a_frag(Alds, P, lane, s, a);
    Frag b0; load_b_frag(Wp, wave * 2 + 0, s, lane, b0);
    Frag b1; load_b_frag(Wp, wave * 2 + 1, s, lane, b1);
    acc0 = wmma_bf16(a, b0, acc0);
    acc1 = wmma_bf16(a, b1, acc1);
  }
#pragma unroll
  for (int r = 0; r < 8; ++r) {                 // C: VGPR r -> M = r + 8*half, N = lane%16
    long a0 = (bond0 + half * 8 + r) * HID + (wave * 2) * 16 + (lane & 15);
    long a1 = a0 + 16;
    float v0 = acc0[r], v1 = acc1[r];
    inp[a0] = v0;               inp[a1] = v1;
    msg[a0] = fmaxf(v0, 0.f);   msg[a1] = fmaxf(v1, 0.f);
  }
}

// ------ GEMM 2: msg_out = relu(inp + (a_msg[b2a] - msg_in[b2revb]) @ W_h)  (K=256) -------
__global__ __launch_bounds__(256) void k_gemm_msg(const float* __restrict__ amsg,
    const float* __restrict__ msg_in, const float* __restrict__ inp,
    const int* __restrict__ b2a, const int* __restrict__ b2revb,
    const unsigned int* __restrict__ Wp, float* __restrict__ msg_out) {
  constexpr int P = 132;                        // 256/2 + 4
  __shared__ unsigned int Alds[16 * P];
  const int t = threadIdx.x, row = t >> 4, kb = t & 15;
  const long bond0 = (long)blockIdx.x * 16;
  const float2* pa = (const float2*)(amsg   + (long)b2a[bond0 + row]    * HID);
  const float2* pm = (const float2*)(msg_in + (long)b2revb[bond0 + row] * HID);
#pragma unroll
  for (int i = 0; i < 8; ++i) {                 // fused gather + subtract into A tile
    int d = kb + i * 16;
    float2 x = pa[d], y = pm[d];
    Alds[row * P + d] = pack2bf(x.x - y.x, x.y - y.y);
  }
  __syncthreads();
  const int wave = t >> 5, lane = t & 31, half = lane >> 4;
  v8f acc0 = {0.f,0.f,0.f,0.f,0.f,0.f,0.f,0.f};
  v8f acc1 = acc0;
#pragma unroll
  for (int s = 0; s < 8; ++s) {
    Frag a;  load_a_frag(Alds, P, lane, s, a);
    Frag b0; load_b_frag(Wp, wave * 2 + 0, s, lane, b0);
    Frag b1; load_b_frag(Wp, wave * 2 + 1, s, lane, b1);
    acc0 = wmma_bf16(a, b0, acc0);
    acc1 = wmma_bf16(a, b1, acc1);
  }
#pragma unroll
  for (int r = 0; r < 8; ++r) {
    long a0 = (bond0 + half * 8 + r) * HID + (wave * 2) * 16 + (lane & 15);
    long a1 = a0 + 16;
    msg_out[a0] = fmaxf(inp[a0] + acc0[r], 0.f);
    msg_out[a1] = fmaxf(inp[a1] + acc1[r], 0.f);
  }
}

// ------ GEMM 3: ah = relu([f_atoms | a_msg] @ W_o + b_o)  (K 389 -> 416) -----------------
__global__ __launch_bounds__(256) void k_gemm_out(const float* __restrict__ f_atoms,
    const float* __restrict__ amsg, const unsigned int* __restrict__ Wp,
    const float* __restrict__ b_o, float* __restrict__ ah) {
  constexpr int KP = 416, P = KP / 2 + 4;       // 212
  __shared__ unsigned int Alds[16 * P];
  const int t = threadIdx.x, row = t >> 4, kb = t & 15;
  const long atom0 = (long)blockIdx.x * 16;
  const float* fa = f_atoms + (atom0 + row) * AFD;
  const float* am = amsg + (atom0 + row) * HID;
#pragma unroll
  for (int i = 0; i < KP / 32; ++i) {           // fused concat into A tile
    int d = kb + i * 16, k = 2 * d;
    float x0 = (k     < AFD) ? fa[k]     : ((k     < OFD) ? am[k     - AFD] : 0.f);
    float x1 = (k + 1 < AFD) ? fa[k + 1] : ((k + 1 < OFD) ? am[k + 1 - AFD] : 0.f);
    Alds[row * P + d] = pack2bf(x0, x1);
  }
  __syncthreads();
  const int wave = t >> 5, lane = t & 31, half = lane >> 4;
  v8f acc0 = {0.f,0.f,0.f,0.f,0.f,0.f,0.f,0.f};
  v8f acc1 = acc0;
#pragma unroll
  for (int s = 0; s < KP / 32; ++s) {
    Frag a;  load_a_frag(Alds, P, lane, s, a);
    Frag b0; load_b_frag(Wp, wave * 2 + 0, s, lane, b0);
    Frag b1; load_b_frag(Wp, wave * 2 + 1, s, lane, b1);
    acc0 = wmma_bf16(a, b0, acc0);
    acc1 = wmma_bf16(a, b1, acc1);
  }
  const int n0 = (wave * 2) * 16 + (lane & 15);
  const float bo0 = b_o[n0], bo1 = b_o[n0 + 16];
#pragma unroll
  for (int r = 0; r < 8; ++r) {
    long a0 = (atom0 + half * 8 + r) * HID + n0;
    ah[a0]      = fmaxf(acc0[r] + bo0, 0.f);
    ah[a0 + 16] = fmaxf(acc1[r] + bo1, 0.f);
  }
}

// ------ per-atom weighted neighbor-message reduction (float4 / b128 path) ----------------
__global__ __launch_bounds__(256) void k_gather(const float* __restrict__ msg,
    const int* __restrict__ a2b, const float* __restrict__ w_bonds,
    float* __restrict__ amsg) {
  const int t = threadIdx.x;
  const int a = blockIdx.x * 4 + (t >> 6);      // 4 atoms per block
  const int h = t & 63;                         // 64 float4 per 256-wide row
  const float4* m4 = (const float4*)msg;
  int bi[MAXNB];
#pragma unroll
  for (int j = 0; j < MAXNB; ++j) {
    bi[j] = a2b[a * MAXNB + j];
    __builtin_prefetch(&m4[(long)bi[j] * 64 + h], 0, 1);
  }
  float4 s = {0.f, 0.f, 0.f, 0.f};
#pragma unroll
  for (int j = 0; j < MAXNB; ++j) {
    float w = w_bonds[bi[j]];
    float4 v = m4[(long)bi[j] * 64 + h];
    s.x += v.x * w; s.y += v.y * w; s.z += v.z * w; s.w += v.w * w;
  }
  ((float4*)amsg)[(long)a * 64 + h] = s;
}

// ---------------- per-molecule segment reduction (mol_ids sorted) ------------------------
__global__ __launch_bounds__(256) void k_zero(float* __restrict__ p, int n) {
  int i = blockIdx.x * 256 + threadIdx.x;
  if (i < n) p[i] = 0.f;
}

__global__ __launch_bounds__(256) void k_segreduce(const float* __restrict__ ah,
    const float* __restrict__ w_atoms, const int* __restrict__ mol_ids,
    float* __restrict__ mol_sum, float* __restrict__ w_sum) {
  const int base = blockIdx.x * 128;
  if (base >= NATOMS) return;
  const int cnt = (NATOMS - base < 128) ? (NATOMS - base) : 128;
  const int t = threadIdx.x;
  int cur = mol_ids[base];
  float acc = 0.f, wacc = 0.f;
  for (int i = 0; i < cnt; ++i) {
    int a = base + i;
    int m = mol_ids[a];
    if (m != cur) {                              // run boundary: flush
      atomicAdd(&mol_sum[(long)cur * HID + t], acc);
      if (t == 0) atomicAdd(&w_sum[cur], wacc);
      acc = 0.f; wacc = 0.f; cur = m;
    }
    float w = w_atoms[a];
    acc += ah[(long)a * HID + t] * w;
    if (t == 0) wacc += w;
  }
  atomicAdd(&mol_sum[(long)cur * HID + t], acc);
  if (t == 0) atomicAdd(&w_sum[cur], wacc);
}

__global__ __launch_bounds__(256) void k_final(const float* __restrict__ mol_sum,
    const float* __restrict__ w_sum, const float* __restrict__ deg,
    float* __restrict__ out) {
  int idx = blockIdx.x * 256 + threadIdx.x;
  if (idx >= NMOLS * HID) return;
  int m = idx / HID;
  float ws = w_sum[m];
  out[idx] = (ws > 0.f) ? deg[m] * mol_sum[idx] / ws : 0.f;
}

// ----------------------------------------------------------------------------------------
extern "C" void kernel_launch(void* const* d_in, const int* in_sizes, int n_in,
                              void* d_out, int out_size, void* d_ws, size_t ws_size,
                              hipStream_t stream) {
  const float* f_atoms = (const float*)d_in[0];
  const float* f_bonds = (const float*)d_in[1];
  const float* w_atoms = (const float*)d_in[2];
  const float* w_bonds = (const float*)d_in[3];
  const float* W_i     = (const float*)d_in[4];
  const float* W_h     = (const float*)d_in[5];
  const float* W_o     = (const float*)d_in[6];
  const float* b_o     = (const float*)d_in[7];
  const float* deg     = (const float*)d_in[8];
  const int*   a2b     = (const int*)d_in[9];
  const int*   b2a     = (const int*)d_in[10];
  const int*   b2revb  = (const int*)d_in[11];
  const int*   mol_ids = (const int*)d_in[12];
  float* out = (float*)d_out;

  char* ws = (char*)d_ws;
  size_t off = 0;
  float* inp  = (float*)(ws + off); off += (size_t)NBONDS * HID * 4;
  float* msgA = (float*)(ws + off); off += (size_t)NBONDS * HID * 4;
  float* msgB = (float*)(ws + off); off += (size_t)NBONDS * HID * 4;
  float* amsg = (float*)(ws + off); off += (size_t)NATOMS * HID * 4;
  unsigned int* WiP = (unsigned int*)(ws + off); off += (size_t)5  * 4096 * 4;
  unsigned int* WhP = (unsigned int*)(ws + off); off += (size_t)8  * 4096 * 4;
  unsigned int* WoP = (unsigned int*)(ws + off); off += (size_t)13 * 4096 * 4;
  float* mol_sum = (float*)(ws + off); off += (size_t)NMOLS * HID * 4;
  float* w_sum   = (float*)(ws + off); off += (size_t)NMOLS * 4;
  float* ah = msgB;                      // reuse: msgB dead after last k_gemm_msg

  // pack weights into WMMA B-fragment layout (bf16)
  k_pack<<<(5 * 4096 + 255) / 256, 256, 0, stream>>>(W_i, BFD, 5 * 4096, WiP);
  k_pack<<<(8 * 4096 + 255) / 256, 256, 0, stream>>>(W_h, HID, 8 * 4096, WhP);
  k_pack<<<(13 * 4096 + 255) / 256, 256, 0, stream>>>(W_o, OFD, 13 * 4096, WoP);

  // input layer
  k_gemm_inp<<<NBONDS / 16, 256, 0, stream>>>(f_bonds, WiP, inp, msgA);

  // DEPTH-1 = 2 message-passing iterations (ping-pong msgA/msgB)
  k_gather<<<NATOMS / 4, 256, 0, stream>>>(msgA, a2b, w_bonds, amsg);
  k_gemm_msg<<<NBONDS / 16, 256, 0, stream>>>(amsg, msgA, inp, b2a, b2revb, WhP, msgB);
  k_gather<<<NATOMS / 4, 256, 0, stream>>>(msgB, a2b, w_bonds, amsg);
  k_gemm_msg<<<NBONDS / 16, 256, 0, stream>>>(amsg, msgB, inp, b2a, b2revb, WhP, msgA);

  // final readout
  k_gather<<<NATOMS / 4, 256, 0, stream>>>(msgA, a2b, w_bonds, amsg);
  k_gemm_out<<<NATOMS / 16, 256, 0, stream>>>(f_atoms, amsg, WoP, b_o, ah);

  // per-molecule weighted mean + degree-of-polymerization scale
  k_zero<<<(NMOLS * HID + NMOLS + 255) / 256, 256, 0, stream>>>(mol_sum, NMOLS * HID + NMOLS);
  k_segreduce<<<(NATOMS + 127) / 128, 256, 0, stream>>>(ah, w_atoms, mol_ids, mol_sum, w_sum);
  k_final<<<(NMOLS * HID + 255) / 256, 256, 0, stream>>>(mol_sum, w_sum, deg, out);
}
